// CantorGlobalAttention_57174604644750
// MI455X (gfx1250) — matrix operand order repaired and checked by reading, not compiled, not run on hardware
//
#include <hip/hip_runtime.h>
#include <math.h>

typedef _Float16 v16h __attribute__((ext_vector_type(16)));
typedef _Float16 v8h  __attribute__((ext_vector_type(8)));
typedef float    v8f  __attribute__((ext_vector_type(8)));
typedef float    v4f  __attribute__((ext_vector_type(4)));

#define NDIR   5
#define E_     16
#define W_     3
#define D_     128
#define P_     256
#define B_     4
#define KTOT   768     // W_*P_
#define KSTEP  32
#define NSTEP  24      // KTOT/KSTEP
#define VT_STRIDE 40   // padded f16 stride per f-row (80B, 16B aligned)

__global__ __launch_bounds__(256) void cantor_attn_kernel(
    const float* __restrict__ Qa,    // (NDIR,E,B,P)
    const float* __restrict__ Ka,    // (NDIR,E,B,P)
    const float* __restrict__ Vg,    // (NDIR,E,B,P,D)
    const float* __restrict__ betas, // (E,E)
    const float* __restrict__ temperature, // (1,)
    const float* __restrict__ fw,    // (NDIR,)
    float* __restrict__ ws)          // (NDIR,B,E,P,D)
{
    __shared__ float    s_s[KTOT];
    __shared__ float    s_q[P_];
    __shared__ float    s_zs[P_];          // wts[x]/Z_p (written post-loop)
    __shared__ _Float16 s_vt[D_ * VT_STRIDE];
    __shared__ float    s_red[16];
    __shared__ int      s_route[W_];
    __shared__ float    s_betaf[W_];
    __shared__ float    s_misc[2];         // invtemp, wts[x]

    const int tid = threadIdx.x;
    const int blk = blockIdx.x;
    const int x  = blk / (E_ * B_);
    const int eb = blk % (E_ * B_);
    const int e  = eb / B_;
    const int b  = eb % B_;

    if (tid == 0) {
        // ---- Cantor coordinates (DEPTH=8) ----
        float coord[E_];
        for (int i = 0; i < E_; ++i) {
            float xx = (float)i / (float)(E_ - 1);
            xx = fmaxf(1e-6f, fminf(xx, 1.0f - 1e-6f));
            float val = 0.f, factor = 0.5f;
            for (int d = 0; d < 8; ++d) {
                xx *= 3.f;
                int dig = (int)xx;
                xx -= (float)dig;
                if (dig == 2) val += factor;
                factor *= 0.5f;
            }
            coord[i] = val;
        }
        // ---- routes: 3 nearest by (dist, index) stable, then sorted ascending ----
        float dref[E_]; bool used[E_];
        for (int i = 0; i < E_; ++i) { dref[i] = fabsf(coord[i] - coord[e]); used[i] = false; }
        int r[W_];
        for (int sel = 0; sel < W_; ++sel) {
            int best = -1;
            for (int i = 0; i < E_; ++i)
                if (!used[i] && (best < 0 || dref[i] < dref[best])) best = i;
            used[best] = true;
            r[sel] = best;
        }
        for (int i = 0; i < W_; ++i)
            for (int j = i + 1; j < W_; ++j)
                if (r[j] < r[i]) { int t = r[i]; r[i] = r[j]; r[j] = t; }
        for (int w = 0; w < W_; ++w) {
            s_route[w] = r[w];
            s_betaf[w] = (r[w] == e) ? 1.0f
                         : 1.0f / (1.0f + __expf(-betas[e * E_ + r[w]]));
        }
        float temp = fabsf(temperature[0]) + 1e-6f;
        s_misc[0] = 1.0f / temp;
        // fusion-weight softmax -> wts[x]
        float mx = fw[0];
        for (int j = 1; j < NDIR; ++j) mx = fmaxf(mx, fw[j]);
        float sm = 0.f;
        for (int j = 0; j < NDIR; ++j) sm += __expf(fw[j] - mx);
        s_misc[1] = __expf(fw[x] - mx) / sm;
    }
    __syncthreads();

    const float invtemp = s_misc[0];
    const float wtsx    = s_misc[1];

    // ---- build q and s in LDS; track local max/min of s ----
    s_q[tid] = Qa[((x * E_ + e) * B_ + b) * P_ + tid];
    float lmax = -INFINITY, lmin = INFINITY;
    #pragma unroll
    for (int j = 0; j < W_; ++j) {
        int wk = tid + j * P_;
        int w  = wk >> 8;
        int k  = wk & 255;
        float sv = Ka[((x * E_ + s_route[w]) * B_ + b) * P_ + k] * s_betaf[w] * invtemp;
        s_s[wk] = sv;
        lmax = fmaxf(lmax, sv);
        lmin = fminf(lmin, sv);
    }
    #pragma unroll
    for (int off = 16; off > 0; off >>= 1) {
        lmax = fmaxf(lmax, __shfl_xor(lmax, off, 32));
        lmin = fminf(lmin, __shfl_xor(lmin, off, 32));
    }
    const int wave = tid >> 5;
    const int lane = tid & 31;
    if (lane == 0) { s_red[wave] = lmax; s_red[8 + wave] = lmin; }
    __syncthreads();
    float smax = s_red[0], smin = s_red[8];
    #pragma unroll
    for (int i = 1; i < 8; ++i) {
        smax = fmaxf(smax, s_red[i]);
        smin = fminf(smin, s_red[8 + i]);
    }

    // ---- main K loop, software-pipelined staging ----
    const int  mrow = lane & 15;
    const bool hi   = lane >= 16;
    const int  kb1  = hi ? 8 : 0;   // A: first K octet this lane holds
    const int  kb2  = kb1 + 16;     // A: second K octet

    // 9th N-tile B matrix: ones in column 0 -> C[:,0] accumulates row sums (Z)
    v16h bones;
    #pragma unroll
    for (int j = 0; j < 16; ++j)
        bones[j] = (mrow == 0) ? (_Float16)1.0f : (_Float16)0.0f;

    v8f acc[2][9];
    #pragma unroll
    for (int mt = 0; mt < 2; ++mt)
        #pragma unroll
        for (int nt = 0; nt < 9; ++nt)
            acc[mt][nt] = (v8f)0.0f;

    // per-thread staging geometry: one contiguous 64B run of V per step
    const int skk = tid >> 3;          // 0..31  (K row within tile)
    const int sf0 = (tid & 7) * 16;    // 0..112 (first f of the run)

    // prefetch tile 0
    v4f pre0, pre1, pre2, pre3;
    {
        const long vb = ((long)(x * E_ + s_route[0]) * B_ + b) * (long)(P_ * D_);
        const v4f* src = (const v4f*)(Vg + vb + (long)(skk & 255) * D_ + sf0);
        pre0 = src[0]; pre1 = src[1]; pre2 = src[2]; pre3 = src[3];
    }

    for (int ks = 0; ks < NSTEP; ++ks) {
        const int k0 = ks * KSTEP;

        __syncthreads();  // previous step's B reads of s_vt are done
        // store prefetched tile (f32 -> f16, transposed to [f][kk])
        {
            _Float16* dst = &s_vt[sf0 * VT_STRIDE + skk];
            dst[ 0 * VT_STRIDE] = (_Float16)pre0[0];
            dst[ 1 * VT_STRIDE] = (_Float16)pre0[1];
            dst[ 2 * VT_STRIDE] = (_Float16)pre0[2];
            dst[ 3 * VT_STRIDE] = (_Float16)pre0[3];
            dst[ 4 * VT_STRIDE] = (_Float16)pre1[0];
            dst[ 5 * VT_STRIDE] = (_Float16)pre1[1];
            dst[ 6 * VT_STRIDE] = (_Float16)pre1[2];
            dst[ 7 * VT_STRIDE] = (_Float16)pre1[3];
            dst[ 8 * VT_STRIDE] = (_Float16)pre2[0];
            dst[ 9 * VT_STRIDE] = (_Float16)pre2[1];
            dst[10 * VT_STRIDE] = (_Float16)pre2[2];
            dst[11 * VT_STRIDE] = (_Float16)pre2[3];
            dst[12 * VT_STRIDE] = (_Float16)pre3[0];
            dst[13 * VT_STRIDE] = (_Float16)pre3[1];
            dst[14 * VT_STRIDE] = (_Float16)pre3[2];
            dst[15 * VT_STRIDE] = (_Float16)pre3[3];
        }
        // issue next tile's global loads (landing under the compute phase)
        if (ks + 1 < NSTEP) {
            const int nk0 = k0 + KSTEP;
            const int nw  = nk0 >> 8;
            const long vb = ((long)(x * E_ + s_route[nw]) * B_ + b) * (long)(P_ * D_);
            const v4f* src =
                (const v4f*)(Vg + vb + (long)((nk0 + skk) & 255) * D_ + sf0);
            pre0 = src[0]; pre1 = src[1]; pre2 = src[2]; pre3 = src[3];
        }
        __syncthreads();

        // build 2 A tiles (16x32 f16) on the fly; normalization deferred
        v16h amat[2];
        #pragma unroll
        for (int mt = 0; mt < 2; ++mt) {
            int   p  = wave * 32 + mt * 16 + mrow;
            float qv = s_q[p];
            float m  = (qv >= 0.f) ? qv * smax : qv * smin;
            v16h a;
            #pragma unroll
            for (int j = 0; j < 8; ++j)
                a[j]     = (_Float16)__expf(qv * s_s[k0 + kb1 + j] - m);
            #pragma unroll
            for (int j = 0; j < 8; ++j)
                a[8 + j] = (_Float16)__expf(qv * s_s[k0 + kb2 + j] - m);
            amat[mt] = a;
        }

        // 8 N tiles of V + 1 ones-tile (row sums)
        #pragma unroll
        for (int nt = 0; nt < 8; ++nt) {
            int f = nt * 16 + mrow;             // this lane's B column
            const _Float16* bp = &s_vt[f * VT_STRIDE + (hi ? 8 : 0)];
            v8h blo = *(const v8h*)(bp);
            v8h bhi = *(const v8h*)(bp + 16);
            v16h bb;
            #pragma unroll
            for (int j = 0; j < 8; ++j) { bb[j] = blo[j]; bb[8 + j] = bhi[j]; }
            acc[0][nt] = __builtin_amdgcn_wmma_f32_16x16x32_f16(
                false, amat[0], false, bb, (short)0, acc[0][nt], false, false);
            acc[1][nt] = __builtin_amdgcn_wmma_f32_16x16x32_f16(
                false, amat[1], false, bb, (short)0, acc[1][nt], false, false);
        }
        acc[0][8] = __builtin_amdgcn_wmma_f32_16x16x32_f16(
            false, amat[0], false, bones, (short)0, acc[0][8], false, false);
        acc[1][8] = __builtin_amdgcn_wmma_f32_16x16x32_f16(
            false, amat[1], false, bones, (short)0, acc[1][8], false, false);
    }

    // ---- publish wts[x]/Z_p from the ones-column (N=0 lives in lanes 0/16) ----
    if (mrow == 0) {
        #pragma unroll
        for (int mt = 0; mt < 2; ++mt)
            #pragma unroll
            for (int j = 0; j < 8; ++j) {
                int p = wave * 32 + mt * 16 + (hi ? 8 + j : j);
                s_zs[p] = wtsx / acc[mt][8][j];
            }
    }
    __syncthreads();

    // ---- scale rows and store this (x,e,b) slice ----
    const long wsbase = (((long)(x * B_ + b) * E_ + e) * P_) * D_;
    #pragma unroll
    for (int mt = 0; mt < 2; ++mt) {
        #pragma unroll
        for (int j = 0; j < 8; ++j) {
            int   p     = wave * 32 + mt * 16 + (hi ? 8 + j : j);
            float scale = s_zs[p];
            #pragma unroll
            for (int nt = 0; nt < 8; ++nt) {
                ws[wsbase + (long)p * D_ + nt * 16 + mrow] = acc[mt][nt][j] * scale;
            }
        }
    }
}

// Sum the 5 direction slices (fusion weights already folded into Z scale).
__global__ __launch_bounds__(256) void cantor_fuse_kernel(
    const float* __restrict__ ws, float* __restrict__ out)
{
    const long n4 = (long)B_ * E_ * P_ * D_ / 4;  // v4f elements per slice
    long i = (long)blockIdx.x * blockDim.x + threadIdx.x;
    if (i >= n4) return;
    const v4f* w4 = (const v4f*)ws;
    v4f a = __builtin_nontemporal_load(&w4[i]);
    #pragma unroll
    for (int xx = 1; xx < NDIR; ++xx) {
        v4f v = __builtin_nontemporal_load(&w4[(long)xx * n4 + i]);
        a = a + v;
    }
    __builtin_nontemporal_store(a, &((v4f*)out)[i]);
}

extern "C" void kernel_launch(void* const* d_in, const int* in_sizes, int n_in,
                              void* d_out, int out_size, void* d_ws, size_t ws_size,
                              hipStream_t stream) {
    const float* Qa    = (const float*)d_in[0];
    const float* Ka    = (const float*)d_in[1];
    const float* Vg    = (const float*)d_in[2];
    const float* betas = (const float*)d_in[3];
    const float* temp  = (const float*)d_in[4];
    const float* fw    = (const float*)d_in[5];
    float* out = (float*)d_out;
    float* ws  = (float*)d_ws;   // needs NDIR*B*E*P*D*4 = 40 MiB

    cantor_attn_kernel<<<NDIR * E_ * B_, 256, 0, stream>>>(
        Qa, Ka, Vg, betas, temp, fw, ws);

    const long n4 = (long)B_ * E_ * P_ * D_ / 4;
    int blocks = (int)((n4 + 255) / 256);
    cantor_fuse_kernel<<<blocks, 256, 0, stream>>>(ws, out);
}
